// SATGiniLoss_23502061044472
// MI455X (gfx1250) — matrix operand order repaired and checked by reading, not compiled, not run on hardware
//
#include <hip/hip_runtime.h>
#include <hip/hip_bf16.h>

typedef __attribute__((ext_vector_type(16))) _Float16 v16h;
typedef __attribute__((ext_vector_type(8)))  float    v8f;

#define NN   2048
#define BB   32
#define CC   10
#define BETA 0.5f

// ---------------------------------------------------------------------------
// Kernel 1: per batch row, x = log(att + 1e-6); bitonic-sort the 2048 values
// ascending in LDS; write sorted row transposed to ws: xs_t[k*32 + b].
// One block (1024 threads) per row.
// ---------------------------------------------------------------------------
__global__ void gini_sort_kernel(const float* __restrict__ att,
                                 float* __restrict__ xs_t) {
    __shared__ float s[NN];
    const int b   = blockIdx.x;
    const int tid = threadIdx.x;

    for (int i = tid; i < NN; i += 1024)
        s[i] = logf(att[b * NN + i] + 1e-6f);
    __syncthreads();

    // Bitonic sort, ascending. Pairs (i, i^j) with (i & j)==0; disjoint per step.
    for (int k = 2; k <= NN; k <<= 1) {
        for (int j = k >> 1; j > 0; j >>= 1) {
            for (int i = tid; i < NN; i += 1024) {
                int l = i ^ j;
                if (l > i) {
                    float a0 = s[i], a1 = s[l];
                    bool up = ((i & k) == 0);
                    if ((a0 > a1) == up) { s[i] = a1; s[l] = a0; }
                }
            }
            __syncthreads();
        }
    }

    for (int i = tid; i < NN; i += 1024)
        xs_t[i * BB + b] = s[i];
}

// ---------------------------------------------------------------------------
// Kernel 2: T_b = sum_k (2k - 2047) * xs_sorted[b][k] via WMMA f16 with a
// hi/lo split (f32-grade accuracy), then cross-entropy and final combine.
// 1 block, 64 threads = 2 waves; wave w handles batch rows 16w .. 16w+15.
// Weights are odd integers <= 2047: exact in f16, and the per-chunk weight
// increment (+64) is also exact -> B fragment updated with one packed add.
// ---------------------------------------------------------------------------
__global__ void gini_ce_combine_kernel(const float* __restrict__ xs_t,
                                       const float* __restrict__ logits,
                                       const int*   __restrict__ labels,
                                       float* __restrict__ out) {
    __shared__ float Tsh[BB];
    __shared__ float ceSh[BB];

    const int tid  = threadIdx.x;
    const int lane = tid & 31;
    const int m0   = (tid >> 5) * 16;     // wave's row tile
    const int h    = (lane >> 4) & 1;     // lane half
    const int M    = m0 + (lane & 15);    // batch row this lane loads for A

    // Build initial B fragment (weights for chunk kb=0) and the +64 splat.
    // B-matrix 32x16 (16-bit) layout: v<4: K=2v+e+8h; v>=4: K=2v+8+e+8h;
    // weight broadcast across all N columns -> value depends on K only.
    v16h bw, binc;
#pragma unroll
    for (int v = 0; v < 8; ++v) {
#pragma unroll
        for (int e = 0; e < 2; ++e) {
            int kbrow = ((v < 4) ? (2 * v) : (2 * v + 8)) + e + 8 * h;
            bw[2 * v + e]   = (_Float16)(float)(2 * kbrow - (NN - 1));
            binc[2 * v + e] = (_Float16)64.0f;
        }
    }

    v8f acc_hi = {};
    v8f acc_lo = {};

    for (int kb = 0; kb < NN; kb += 32) {
        // Prefetch next chunk's A-fragment region (gfx1250 global_prefetch).
        if (kb + 32 < NN)
            __builtin_prefetch(&xs_t[(kb + 32) * BB + M], 0, 3);

        v16h ahi, alo;
#pragma unroll
        for (int v = 0; v < 8; ++v) {
#pragma unroll
            for (int e = 0; e < 2; ++e) {
                // A-matrix 16-bit 16x32 layout (ISA 7.12.2):
                //   v<4 : K = 2v+e + 8h ;  v>=4 : K = 2v+8 + e + 8h
                int ka = ((v < 4) ? (2 * v) : (2 * v + 8)) + e + 8 * h;
                float x = xs_t[(kb + ka) * BB + M];
                _Float16 hi = (_Float16)x;
                ahi[2 * v + e] = hi;
                alo[2 * v + e] = (_Float16)(x - (float)hi);
            }
        }
        // Independent accumulators: the two WMMAs per chunk can pipeline.
        acc_hi = __builtin_amdgcn_wmma_f32_16x16x32_f16(
                     false, ahi, false, bw, (short)0, acc_hi, false, false);
        acc_lo = __builtin_amdgcn_wmma_f32_16x16x32_f16(
                     false, alo, false, bw, (short)0, acc_lo, false, false);
        bw = bw + binc;   // exact f16 update: lowers to v_pk_add_f16
    }

    // D layout (16x16 f32): VGPR v: lanes 0-15 -> M=v, lanes 16-31 -> M=v+8,
    // column N = lane&15; all columns identical here, take column 0.
    if ((lane & 15) == 0) {
#pragma unroll
        for (int v = 0; v < 8; ++v)
            Tsh[m0 + 8 * h + v] = acc_hi[v] + acc_lo[v];
    }
    __syncthreads();

    // Cross-entropy: one thread per batch row.
    if (tid < BB) {
        const float* lg = logits + tid * CC;
        float m = lg[0];
#pragma unroll
        for (int c = 1; c < CC; ++c) m = fmaxf(m, lg[c]);
        float sum = 0.0f;
#pragma unroll
        for (int c = 0; c < CC; ++c) sum += expf(lg[c] - m);
        float lse = m + logf(sum);
        ceSh[tid] = lse - lg[labels[tid]];
    }
    __syncthreads();

    if (tid == 0) {
        float ce = 0.0f;
        for (int b = 0; b < BB; ++b) ce += ceSh[b];
        ce *= (1.0f / (float)BB);

        const float denom = 2.0f * (float)NN * (float)NN + 1e-9f;
        float info = 0.0f;
        for (int b = 0; b < BB; ++b) info += (2.0f * Tsh[b]) / denom;

        out[0] = ce - BETA * info;
    }
}

extern "C" void kernel_launch(void* const* d_in, const int* in_sizes, int n_in,
                              void* d_out, int out_size, void* d_ws, size_t ws_size,
                              hipStream_t stream) {
    const float* att    = (const float*)d_in[0];   // [32, 2048] f32
    const float* logits = (const float*)d_in[1];   // [32, 10]   f32
    const int*   labels = (const int*)  d_in[2];   // [32]       i32
    float*       out    = (float*)d_out;           // scalar f32
    float*       xs_t   = (float*)d_ws;            // [2048, 32] f32 = 256 KB

    gini_sort_kernel<<<BB, 1024, 0, stream>>>(att, xs_t);
    gini_ce_combine_kernel<<<1, 64, 0, stream>>>(xs_t, logits, labels, out);
}
